// DynamicDepthSeparableTimeSeriesSelfAttention_1537598292709
// MI455X (gfx1250) — matrix operand order repaired and checked by reading, not compiled, not run on hardware
//
#include <hip/hip_runtime.h>
#include <hip/hip_bf16.h>
#include <math.h>

// ---------------- CDNA5 WMMA types ----------------
typedef _Float16 v16h __attribute__((ext_vector_type(16)));
typedef float    v8f  __attribute__((ext_vector_type(8)));
typedef int      g4i  __attribute__((__vector_size__(16)));  // matches builtin param type

struct alignas(16) H8 { _Float16 h[8]; };

// CDNA5 async global->LDS DMA (ASYNCcnt path), with graceful fallback.
// Probe result: signature is (g4i* src, AS3 g4i* lds_dst, imm offset, imm cpol).
#if defined(__AMDGCN__) && __has_builtin(__builtin_amdgcn_global_load_async_to_lds_b128)
#define HAVE_ASYNC_LDS 1
// AS3 value == low 32 bits of flat LDS addr (aperture mapping keeps offset in [31:0])
#define AS3P(p) ((__attribute__((address_space(3))) g4i*)(unsigned)(unsigned long long)(p))
#endif

__device__ __forceinline__ void wait_async_then_barrier() {
#if defined(HAVE_ASYNC_LDS)
#if __has_builtin(__builtin_amdgcn_s_wait_asynccnt)
    __builtin_amdgcn_s_wait_asynccnt(0);
#else
    asm volatile("s_wait_asynccnt 0x0" ::: "memory");
#endif
#endif
    __syncthreads();
}

// A-matrix fragment (16x32 f16, M x K): lane m = L%16,
// element e -> k = (L/16)*16 + e. Two b128 LDS loads.
__device__ __forceinline__ v16h load_a_frag(const _Float16* row, int h, int f) {
    H8 lo = *(const H8*)(row + f * 32 + h * 16);
    H8 hi = *(const H8*)(row + f * 32 + h * 16 + 8);
    v16h a;
#pragma unroll
    for (int e = 0; e < 8; ++e) { a[e] = lo.h[e]; a[e + 8] = hi.h[e]; }
    return a;
}

// B-matrix fragment (32x16 f16, K x N): lane n = L%16,
// element e -> k = (L/16)*16 + e. 16 contiguous halves.
__device__ __forceinline__ v16h load_b_frag(const _Float16* row, int h, int f) {
    H8 lo = *(const H8*)(row + f * 32 + h * 16);
    H8 hi = *(const H8*)(row + f * 32 + h * 16 + 8);
    v16h b;
#pragma unroll
    for (int e = 0; e < 8; ++e) { b[e] = lo.h[e]; b[e + 8] = hi.h[e]; }
    return b;
}

#define WMMA_F16(A, B, C) \
    __builtin_amdgcn_wmma_f32_16x16x32_f16(false, (A), false, (B), (short)0, (C), false, false)

// ---------------- Kernel 1: fused DDS conv ----------------
// y = pw @ x (pointwise 1x1), then z = g0*conv3(y) + g1*conv15(y).
// mode 0: write q = z * 64^-0.25 as f16 [n][l][c]   (c contiguous)
// mode 1: write v = z           as f16 [n][c][l]   (l contiguous)
#define LT 64
#define NP (LT + 14)
#define PS 82  // padded LDS stride (conflict-free spread across 64 banks)

__global__ __launch_bounds__(256) void dds_conv_kernel(
    const float* __restrict__ x,
    const float* __restrict__ pw_qk, const float* __restrict__ dw3_qk,
    const float* __restrict__ dw15_qk, const float* __restrict__ gate_qk,
    const float* __restrict__ pw_v, const float* __restrict__ dw3_v,
    const float* __restrict__ dw15_v, const float* __restrict__ gate_v,
    _Float16* __restrict__ qg, _Float16* __restrict__ vg) {
    const int lt   = blockIdx.x;        // 0..31 (L tiles of 64)
    const int hh   = blockIdx.y;        // 0..7  head
    const int b    = blockIdx.z >> 1;   // 0..3
    const int mode = blockIdx.z & 1;    // 0 = qk, 1 = v
    const int tid  = threadIdx.x;
    const int l0   = lt * LT;

    const float* pw   = mode ? pw_v   : pw_qk;
    const float* dw3  = mode ? dw3_v  : dw3_qk;
    const float* dw15 = mode ? dw15_v : dw15_qk;
    const float* gate = mode ? gate_v : gate_qk;

    __shared__ float xs[64 * PS];
    __shared__ float ys[64 * PS];
    __shared__ float pwS[64 * 64];
    __shared__ float d3[64 * 3];
    __shared__ float d15[64 * 15];

    // stage x tile with halo of 7, zero padded ('same' conv)
    for (int idx = tid; idx < 64 * NP; idx += 256) {
        int i = idx / NP, p = idx % NP;
        int l = l0 - 7 + p;
        float v = 0.f;
        if (l >= 0 && l < 2048) v = x[((size_t)b * 64 + i) * 2048 + l];
        xs[i * PS + p] = v;
    }
    for (int idx = tid; idx < 64 * 64; idx += 256) pwS[idx] = pw[(size_t)hh * 4096 + idx];
    for (int idx = tid; idx < 64 * 3; idx += 256)  d3[idx]  = dw3[(size_t)hh * 192 + idx];
    for (int idx = tid; idx < 64 * 15; idx += 256) d15[idx] = dw15[(size_t)hh * 960 + idx];
    __syncthreads();

    // pointwise: y[o][p] = sum_i pw[o][i] * x[i][p] (incl. halo)
    for (int idx = tid; idx < 64 * NP; idx += 256) {
        int o = idx / NP, p = idx % NP;
        float acc = 0.f;
        for (int i = 0; i < 64; ++i) acc += pwS[o * 64 + i] * xs[i * PS + p];
        ys[o * PS + p] = acc;
    }
    __syncthreads();

    // gate softmax (2-way)
    float g0 = gate[0], g1 = gate[1];
    float mg = fmaxf(g0, g1);
    float ea = __expf(g0 - mg), eb = __expf(g1 - mg);
    float gs0 = ea / (ea + eb), gs1 = eb / (ea + eb);
    const float scale = 0.35355339059f;  // 64^-0.25
    const int n = b * 8 + hh;

    if (mode == 0) {
        for (int idx = tid; idx < 64 * LT; idx += 256) {
            int o = idx & 63, ll = idx >> 6;
            float a3 = 0.f, a15 = 0.f;
#pragma unroll
            for (int t = 0; t < 3; ++t)  a3  += d3[o * 3 + t]  * ys[o * PS + ll + 6 + t];
#pragma unroll
            for (int t = 0; t < 15; ++t) a15 += d15[o * 15 + t] * ys[o * PS + ll + t];
            float z = gs0 * a3 + gs1 * a15;
            qg[((size_t)n * 2048 + l0 + ll) * 64 + o] = (_Float16)(z * scale);
        }
    } else {
        for (int idx = tid; idx < 64 * LT; idx += 256) {
            int ll = idx & 63, o = idx >> 6;
            float a3 = 0.f, a15 = 0.f;
#pragma unroll
            for (int t = 0; t < 3; ++t)  a3  += d3[o * 3 + t]  * ys[o * PS + ll + 6 + t];
#pragma unroll
            for (int t = 0; t < 15; ++t) a15 += d15[o * 15 + t] * ys[o * PS + ll + t];
            float z = gs0 * a3 + gs1 * a15;
            vg[((size_t)n * 64 + o) * 2048 + l0 + ll] = (_Float16)z;
        }
    }
}

// ---------------- Kernel 2: fused flash attention (WMMA) ----------------
// Per block: one head n, one 128-wide column tile j. 8 waves, wave w owns 16
// columns. Loop over 16 key tiles of 128: S = q_i^T q_j (K=64 via 2x K32
// WMMA), online softmax over keys i (axis=1), O += V * P (K=128 via 4x K32).
// Tile staging uses GLOBAL_LOAD_ASYNC_TO_LDS_B128 (ASYNCcnt) when available.
__global__ __launch_bounds__(256) void flash_attn_kernel(
    const _Float16* __restrict__ qg,   // [32][2048][64]
    const _Float16* __restrict__ vg,   // [32][64][2048]
    float* __restrict__ attn) {        // [32][2048][64]  (softmax-weighted V)
    const int n    = blockIdx.x;
    const int j0   = blockIdx.y * 128;
    const int tid  = threadIdx.x;
    const int lane = tid & 31;
    const int w    = tid >> 5;
    const int ln16 = lane & 15;
    const int h    = lane >> 4;

    __shared__ alignas(16) _Float16 lq[128 * 64];  // [i_local][c]
    __shared__ alignas(16) _Float16 lv[64 * 128];  // [c][i_local]

    // persistent B fragments: q columns of this wave (K = c = 64)
    const _Float16* qrow = qg + ((size_t)n * 2048 + (j0 + w * 16 + ln16)) * 64;
    const v16h qB0 = load_b_frag(qrow, h, 0);
    const v16h qB1 = load_b_frag(qrow, h, 1);

    v8f O[4];
#pragma unroll
    for (int cc = 0; cc < 4; ++cc)
#pragma unroll
        for (int r = 0; r < 8; ++r) O[cc][r] = 0.f;
    float m_run = -1.0e30f, s_run = 0.f;

    const uint4* qg4 = (const uint4*)qg;
    const uint4* vg4 = (const uint4*)vg;
    uint4* lq4 = (uint4*)lq;
    uint4* lv4 = (uint4*)lv;

#pragma unroll 1
    for (int it = 0; it < 16; ++it) {
        const int i0 = it * 128;
        if (it) __syncthreads();
        // stage q tile [128][64] and v tile [64][128] (f16, 16 KB each)
#pragma unroll
        for (int s = 0; s < 4; ++s) {
            int idx = tid + s * 256;
            int qr = idx >> 3, qc = idx & 7;   // 8 uint4 per q row
            const uint4* gq = qg4 + ((size_t)n * 2048 + i0 + qr) * 8 + qc;
            int vr = idx >> 4, vc = idx & 15;  // 16 uint4 per v row
            const uint4* gv = vg4 + ((size_t)n * 64 + vr) * 256 + (i0 >> 3) + vc;
#if defined(HAVE_ASYNC_LDS)
            __builtin_amdgcn_global_load_async_to_lds_b128((g4i*)gq, AS3P(lq4 + idx), 0, 0);
            __builtin_amdgcn_global_load_async_to_lds_b128((g4i*)gv, AS3P(lv4 + idx), 0, 0);
#else
            lq4[idx] = *gq;
            lv4[idx] = *gv;
#endif
        }
        wait_async_then_barrier();

        // ---- S = q_i^T q_j : 8 row tiles x (2 WMMA, K=64) ----
        v8f st[8];
#pragma unroll
        for (int t = 0; t < 8; ++t) {
            const _Float16* ar = lq + (size_t)(t * 16 + ln16) * 64;
            v16h a0 = load_a_frag(ar, h, 0);
            v16h a1 = load_a_frag(ar, h, 1);
            v8f c;
#pragma unroll
            for (int r = 0; r < 8; ++r) c[r] = 0.f;
            c = WMMA_F16(a0, qB0, c);
            c = WMMA_F16(a1, qB1, c);
            st[t] = c;
        }

        // ---- online softmax over key axis (columns of S tile) ----
        float mt = -1.0e30f;
#pragma unroll
        for (int t = 0; t < 8; ++t)
#pragma unroll
            for (int r = 0; r < 8; ++r) mt = fmaxf(mt, st[t][r]);
        mt = fmaxf(mt, __shfl_xor(mt, 16, 32));  // lane pair = same column
        float m_new = fmaxf(m_run, mt);
        float alpha = __expf(m_run - m_new);
        float ssum = 0.f;

        // build P in B-fragment layout; D->B needs only a half-wave exchange
        v16h pF[4];
#pragma unroll
        for (int u = 0; u < 4; ++u) {
            float e0[8], e1[8];
#pragma unroll
            for (int r = 0; r < 8; ++r) {
                e0[r] = __expf(st[2 * u][r] - m_new);
                e1[r] = __expf(st[2 * u + 1][r] - m_new);
                ssum += e0[r] + e1[r];
            }
#pragma unroll
            for (int r = 0; r < 8; ++r) {
                float x0 = __shfl_xor(e0[r], 16, 32);
                float x1 = __shfl_xor(e1[r], 16, 32);
                pF[u][r]     = (_Float16)(h == 0 ? e0[r] : x1);
                pF[u][r + 8] = (_Float16)(h == 0 ? x0 : e1[r]);
            }
        }
        ssum += __shfl_xor(ssum, 16, 32);
        s_run = s_run * alpha + ssum;
        m_run = m_new;

        // rescale accumulators, then O += V * P (K = 128)
#pragma unroll
        for (int cc = 0; cc < 4; ++cc) {
#pragma unroll
            for (int r = 0; r < 8; ++r) O[cc][r] *= alpha;
            const _Float16* vrow = lv + (size_t)(cc * 16 + ln16) * 128;
            v8f acc = O[cc];
#pragma unroll
            for (int u = 0; u < 4; ++u) {
                v16h va = load_a_frag(vrow, h, u);
                acc = WMMA_F16(va, pF[u], acc);
            }
            O[cc] = acc;
        }
    }

    const float inv = 1.0f / s_run;
    const int j = j0 + w * 16 + ln16;
#pragma unroll
    for (int cc = 0; cc < 4; ++cc)
#pragma unroll
        for (int r = 0; r < 8; ++r) {
            int c = cc * 16 + r + 8 * h;
            attn[((size_t)n * 2048 + j) * 64 + c] = O[cc][r] * inv;
        }
}

// ---------------- Kernel 3: unify heads (1x1 conv, f32) ----------------
__global__ __launch_bounds__(256) void unify_kernel(
    const float* __restrict__ unify,   // [64][512]
    const float* __restrict__ attn,    // [32][2048][64]
    float* __restrict__ out) {         // [4][64][2048]
    const int b = blockIdx.y;
    const int l = blockIdx.x * 4 + (threadIdx.x >> 6);
    const int o = threadIdx.x & 63;
    const float* ur = unify + (size_t)o * 512;
    float acc = 0.f;
    for (int hh = 0; hh < 8; ++hh) {
        const float* ar = attn + ((size_t)(b * 8 + hh) * 2048 + l) * 64;
#pragma unroll
        for (int c = 0; c < 64; ++c) acc += ur[hh * 64 + c] * ar[c];
    }
    out[((size_t)b * 64 + o) * 2048 + l] = acc;
}

// ---------------- launcher ----------------
extern "C" void kernel_launch(void* const* d_in, const int* in_sizes, int n_in,
                              void* d_out, int out_size, void* d_ws, size_t ws_size,
                              hipStream_t stream) {
    const float* x       = (const float*)d_in[0];
    const float* pw_qk   = (const float*)d_in[1];
    const float* dw3_qk  = (const float*)d_in[2];
    const float* dw15_qk = (const float*)d_in[3];
    const float* gate_qk = (const float*)d_in[4];
    const float* pw_v    = (const float*)d_in[5];
    const float* dw3_v   = (const float*)d_in[6];
    const float* dw15_v  = (const float*)d_in[7];
    const float* gate_v  = (const float*)d_in[8];
    const float* unify   = (const float*)d_in[9];

    // workspace: q f16 [32][2048][64] | v f16 [32][64][2048] | attn f32 [32][2048][64]
    _Float16* qg = (_Float16*)d_ws;
    _Float16* vg = qg + (size_t)32 * 2048 * 64;
    float* attn  = (float*)(vg + (size_t)32 * 64 * 2048);

    dds_conv_kernel<<<dim3(32, 8, 8), 256, 0, stream>>>(
        x, pw_qk, dw3_qk, dw15_qk, gate_qk, pw_v, dw3_v, dw15_v, gate_v, qg, vg);
    flash_attn_kernel<<<dim3(32, 16), 256, 0, stream>>>(qg, vg, attn);
    unify_kernel<<<dim3(512, 4), 256, 0, stream>>>(unify, attn, (float*)d_out);
}
